// EntityAttentionRNNGATAgent_81784767251104
// MI455X (gfx1250) — compile-verified, matrix operand order checked by reading, not run on hardware
//
#include <hip/hip_runtime.h>

#define BS 32
#define TS 64
#define NE 64
#define ED 96
#define NAg 16
#define EMB 128
#define RH 128
#define NACT 16
#define BSTS (BS*TS)

typedef _Float16 F16;
typedef __attribute__((ext_vector_type(16))) _Float16 v16h;
typedef __attribute__((ext_vector_type(8)))  float    v8f;

#if defined(__has_builtin)
#if __has_builtin(__builtin_amdgcn_global_load_async_to_lds_b128) && \
    __has_builtin(__builtin_amdgcn_s_wait_asynccnt)
#define HAS_ASYNC_LD 1
#endif
#if __has_builtin(__builtin_amdgcn_global_store_async_from_lds_b128) && \
    __has_builtin(__builtin_amdgcn_s_wait_asynccnt)
#define HAS_ASYNC_ST 1
#endif
#endif

// Builtin prototype (from hipcc diagnostic): params are int4* with AS qualifiers
typedef int v4i __attribute__((vector_size(16)));
typedef __attribute__((address_space(1))) v4i* as1_v4i_p;   // global
typedef __attribute__((address_space(3))) v4i* as3_v4i_p;   // LDS

// 16B global -> LDS (async engine when available; tracked on ASYNCcnt)
static __device__ __forceinline__ void g2l_b128(const void* g, void* l) {
#ifdef HAS_ASYNC_LD
  __builtin_amdgcn_global_load_async_to_lds_b128((as1_v4i_p)g, (as3_v4i_p)l, 0, 0);
#else
  *(float4*)l = *(const float4*)g;
#endif
}
// 16B LDS -> global
static __device__ __forceinline__ void l2g_b128(void* g, const void* l) {
#ifdef HAS_ASYNC_ST
  __builtin_amdgcn_global_store_async_from_lds_b128((as1_v4i_p)g, (as3_v4i_p)l, 0, 0);
#else
  *(float4*)g = *(const float4*)l;
#endif
}
static __device__ __forceinline__ void async_wait0() {
#if defined(HAS_ASYNC_LD) || defined(HAS_ASYNC_ST)
  __builtin_amdgcn_s_wait_asynccnt(0);
#endif
}

static __device__ __forceinline__ v8f wmma_f16(v16h a, v16h b, v8f c) {
  // D = A(16x32,f16) * B(32x16,f16) + C(16x16,f32)
  return __builtin_amdgcn_wmma_f32_16x16x32_f16(false, a, false, b, (short)0, c, false, false);
}

// A tile 16x32 from row-major A[M][K] (leading dim ld). ISA 7.12.2 wave32 layout.
static __device__ __forceinline__ v16h load_a16(const F16* base, int ld) {
  int lane = threadIdx.x & 31;
  int m = lane & 15;
  int khi = (lane & 16) ? 8 : 0;
  const F16* row = base + m * ld;
  v16h f;
#pragma unroll
  for (int v = 0; v < 8; ++v) {
    int kb = (v < 4) ? (khi + 2 * v) : (16 + khi + 2 * (v - 4));
    f[2 * v]     = row[kb];
    f[2 * v + 1] = row[kb + 1];
  }
  return f;
}
// B tile 32x16 read from a row-major W[N][K] matrix (B = W^T): elem(k,n)=base[n*ld+k]
static __device__ __forceinline__ v16h load_bT16(const F16* base, int ld) {
  int lane = threadIdx.x & 31;
  int n = lane & 15;
  int khi = (lane & 16) ? 8 : 0;
  const F16* col = base + n * ld;
  v16h f;
#pragma unroll
  for (int v = 0; v < 8; ++v) {
    int kb = (v < 4) ? (khi + 2 * v) : (16 + khi + 2 * (v - 4));
    f[2 * v]     = col[kb];
    f[2 * v + 1] = col[kb + 1];
  }
  return f;
}
// B tile 32x16 from row-major B[K][N]: elem(k,n)=base[k*ld+n]
static __device__ __forceinline__ v16h load_b16(const F16* base, int ld) {
  int lane = threadIdx.x & 31;
  int n = lane & 15;
  int khi = (lane & 16) ? 8 : 0;
  v16h f;
#pragma unroll
  for (int v = 0; v < 8; ++v) {
    int kb = (v < 4) ? (khi + 2 * v) : (16 + khi + 2 * (v - 4));
    f[2 * v]     = base[kb * ld + n];
    f[2 * v + 1] = base[(kb + 1) * ld + n];
  }
  return f;
}

static __device__ __forceinline__ float u01hash(unsigned long long x) {
  x ^= x >> 33; x *= 0xff51afd7ed558ccdULL;
  x ^= x >> 33; x *= 0xc4ceb9fe1a85ec53ULL;
  x ^= x >> 33;
  unsigned r = (unsigned)(x >> 40);
  return (r + 0.5f) * (1.0f / 16777216.0f);   // strictly in (0,1)
}

__global__ void cvt_f32_to_f16(const float* __restrict__ s, F16* __restrict__ d, int nElem) {
  int i = blockIdx.x * 256 + threadIdx.x;
  if (i < nElem) d[i] = (F16)s[i];
}

// ---------------- K1: fused fc1 + 64-entity MHA + Wout + fc2, one block per (b,t) -------------
__global__ void __launch_bounds__(128) k1_fc1_attn_fc2(
    const float* __restrict__ ents, const unsigned char* __restrict__ om,
    const unsigned char* __restrict__ em,
    const F16* __restrict__ wfc1, const float* __restrict__ bfc1,
    const F16* __restrict__ wawin,
    const F16* __restrict__ wawout, const float* __restrict__ baout,
    const F16* __restrict__ wfc2, const float* __restrict__ bfc2,
    F16* __restrict__ x3h)
{
  __shared__ alignas(16) char smem[61440];
  F16*   sX1  = (F16*)smem;               // 64x128 f16
  F16*   sK   = (F16*)(smem + 16384);     // 64x128
  F16*   sV   = (F16*)(smem + 32768);     // 64x128
  F16*   sQ   = (F16*)(smem + 49152);     // 16x128
  F16*   sE   = (F16*)(smem + 16384);     // 64x96 f16 (phase A, aliases sK)
  float* sEf  = (float*)(smem + 28672);   // 64x96 f32 staging (phase A, aliases sV/sQ)
  float* sLf  = (float*)smem;             // 4 heads x 16x64 logits (aliases dead sX1)
  F16*   sLh  = (F16*)(smem + 53248);     // 4 x 16x64 f16 softmax weights
  float* sO   = (float*)smem;             // 16x128 attn out (aliases dead sLf)
  F16*   sO16 = (F16*)(smem + 8192);      // 16x128 (A for Wout; reused as fc2 out tile)
  F16*   sX2h = (F16*)(smem + 12288);     // 16x128

  const int tid = threadIdx.x, lane = tid & 31, wv = tid >> 5;
  const int e = blockIdx.x;
  const int n = lane & 15, mh = (lane & 16) ? 8 : 0;

  // Phase A: async-stage entities (f32) into LDS, then convert to f16
  {
    const char* g = (const char*)(ents + (size_t)e * NE * ED);
    char* l = (char*)sEf;
    for (int c = tid; c < (NE * ED * 4) / 16; c += 128)   // 1536 x 16B
      g2l_b128(g + c * 16, l + c * 16);
    async_wait0();
  }
  __syncthreads();
  for (int i = tid; i < NE * ED; i += 128) sE[i] = (F16)sEf[i];
  __syncthreads();

  // fc1: x1 = relu(E @ W1^T + b1)  (64x128, K=96)
  for (int t = wv; t < 32; t += 4) {
    int m0 = (t >> 3) * 16, n0 = (t & 7) * 16;
    v8f acc = {};
    for (int k0 = 0; k0 < 96; k0 += 32)
      acc = wmma_f16(load_a16(sE + m0 * ED + k0, ED),
                     load_bT16(wfc1 + n0 * ED + k0, ED), acc);
#pragma unroll
    for (int i = 0; i < 8; ++i)
      sX1[(m0 + mh + i) * EMB + n0 + n] = (F16)fmaxf(acc[i] + bfc1[n0 + n], 0.f);
  }
  __syncthreads();

  // qkv = x1 @ Win^T (q rows 0..15 only; k,v: all 64 rows)
  for (int t = wv; t < 72; t += 4) {
    int m0, gn0, dcol; F16* dst;
    if (t < 8) { m0 = 0; gn0 = t * 16; dcol = gn0; dst = sQ; }
    else {
      int u = t - 8, which = u >> 5; u &= 31;
      m0 = (u >> 3) * 16; dcol = (u & 7) * 16;
      gn0 = 128 + which * 128 + dcol;
      dst = which ? sV : sK;
    }
    v8f acc = {};
    for (int k0 = 0; k0 < 128; k0 += 32)
      acc = wmma_f16(load_a16(sX1 + m0 * EMB + k0, EMB),
                     load_bT16(wawin + gn0 * EMB + k0, EMB), acc);
#pragma unroll
    for (int i = 0; i < 8; ++i)
      dst[(m0 + mh + i) * EMB + dcol + n] = (F16)acc[i];
  }
  __syncthreads();

  // logits per head (wave == head): 16x64, K=32 head dim
  {
    const unsigned char* omr = om + (size_t)e * NE * NE;
    int h = wv;
    v16h aq = load_a16(sQ + h * 32, EMB);
    for (int nt = 0; nt < 4; ++nt) {
      int n0 = nt * 16;
      v8f acc = {};
      acc = wmma_f16(aq, load_bT16(sK + n0 * EMB + h * 32, EMB), acc);
#pragma unroll
      for (int i = 0; i < 8; ++i) {
        int m = mh + i, col = n0 + n;
        float v = acc[i] * 0.17677669529663689f;   // 1/sqrt(32)
        if (omr[m * NE + col]) v = -3.0e38f;
        sLf[h * 1024 + m * 64 + col] = v;
      }
    }
  }
  __syncthreads();
  // row softmax (lane m < 16 handles one row; fully-masked rows -> 0)
  if (lane < 16) {
    float* row  = sLf + wv * 1024 + lane * 64;
    F16*   wrow = sLh + wv * 1024 + lane * 64;
    float mx = -3.0e38f;
    for (int j = 0; j < 64; ++j) mx = fmaxf(mx, row[j]);
    if (mx > -1.0e38f) {
      float s = 0.f;
      for (int j = 0; j < 64; ++j) s += expf(row[j] - mx);
      float inv = 1.0f / s;
      for (int j = 0; j < 64; ++j) wrow[j] = (F16)(expf(row[j] - mx) * inv);
    } else {
      for (int j = 0; j < 64; ++j) wrow[j] = (F16)0.0f;
    }
  }
  __syncthreads();
  // out = w @ v (per head, K=64, N=32)
  {
    int h = wv;
    for (int nt = 0; nt < 2; ++nt) {
      int n0 = nt * 16;
      v8f acc = {};
      for (int k0 = 0; k0 < 64; k0 += 32)
        acc = wmma_f16(load_a16(sLh + h * 1024 + k0, 64),
                       load_b16(sV + k0 * EMB + h * 32 + n0, EMB), acc);
#pragma unroll
      for (int i = 0; i < 8; ++i)
        sO[(mh + i) * EMB + h * 32 + n0 + n] = acc[i];
    }
  }
  __syncthreads();
  for (int i = tid; i < NAg * EMB; i += 128) sO16[i] = (F16)sO[i];
  __syncthreads();

  const unsigned char* emr = em + (size_t)e * NE;
  // Wout + bias + post-mask -> x2 (f16)
  for (int nt = wv; nt < 8; nt += 4) {
    int n0 = nt * 16;
    v8f acc = {};
    for (int k0 = 0; k0 < 128; k0 += 32)
      acc = wmma_f16(load_a16(sO16 + k0, EMB),
                     load_bT16(wawout + n0 * EMB + k0, EMB), acc);
#pragma unroll
    for (int i = 0; i < 8; ++i) {
      int m = mh + i;
      float v = acc[i] + baout[n0 + n];
      if (emr[m]) v = 0.f;
      sX2h[m * EMB + n0 + n] = (F16)v;
    }
  }
  __syncthreads();
  // fc2 + relu -> LDS tile (sO16 region reused), then coalesced/async drain to x3h
  for (int nt = wv; nt < 8; nt += 4) {
    int n0 = nt * 16;
    v8f acc = {};
    for (int k0 = 0; k0 < 128; k0 += 32)
      acc = wmma_f16(load_a16(sX2h + k0, EMB),
                     load_bT16(wfc2 + n0 * EMB + k0, EMB), acc);
#pragma unroll
    for (int i = 0; i < 8; ++i)
      sO16[(mh + i) * RH + n0 + n] = (F16)fmaxf(acc[i] + bfc2[n0 + n], 0.f);
  }
  __syncthreads();
  {
    char* g = (char*)(x3h + (size_t)e * NAg * RH);
    const char* l = (const char*)sO16;
    for (int c = tid; c < 256; c += 128)                   // 4KB tile
      l2g_b128(g + c * 16, l + c * 16);
  }
  // implicit S_WAIT_IDLE at S_ENDPGM drains ASYNCcnt
}

// ---------------- K2: persistent GRU scan; one block per batch b; 8 waves; double-buffered x --
__global__ void __launch_bounds__(256) k2_gru(
    const float* __restrict__ h0, const F16* __restrict__ x3h,
    const F16* __restrict__ wih, const F16* __restrict__ whh,
    const float* __restrict__ bih, const float* __restrict__ bhh,
    float* __restrict__ hs_out, F16* __restrict__ hsh)
{
  __shared__ alignas(16) char smem[53248];
  float* sH   = (float*)smem;             // 16x128 f32 state
  F16*   sH16 = (F16*)(smem + 8192);
  F16*   sXa  = (F16*)(smem + 12288);     // x tile buffer A
  float* sRZ  = (float*)(smem + 16384);   // 16x256 (ir+hr | iz+hz)
  float* sIN  = (float*)(smem + 32768);   // 16x128 (inn)
  float* sHN  = (float*)(smem + 40960);   // 16x128 (hn)
  F16*   sXb  = (F16*)(smem + 49152);     // x tile buffer B
  const int tid = threadIdx.x, lane = tid & 31, wv = tid >> 5;
  const int b = blockIdx.x;
  const int n = lane & 15, mh = (lane & 16) ? 8 : 0;

  for (int i = tid; i < NAg * RH; i += 256) {
    float v = h0[(size_t)b * NAg * RH + i];
    sH[i] = v; sH16[i] = (F16)v;
  }
  // preload x tile for t=0 (one 16B chunk per thread; ASYNCcnt-tracked)
  {
    const char* g = (const char*)(x3h + (size_t)(b * TS) * NAg * RH);
    g2l_b128(g + tid * 16, (char*)sXa + tid * 16);
  }

  for (int t = 0; t < TS; ++t) {
    int e = b * TS + t;
    F16* sXt  = (t & 1) ? sXb : sXa;
    F16* sXnx = (t & 1) ? sXa : sXb;
    async_wait0();
    __syncthreads();
    // prefetch x for t+1 into the other buffer while this step computes
    if (t + 1 < TS) {
      const char* g = (const char*)(x3h + (size_t)(e + 1) * NAg * RH);
      g2l_b128(g + tid * 16, (char*)sXnx + tid * 16);
    }
    __builtin_prefetch(wih, 0, 1);

    for (int nt = wv; nt < 24; nt += 8) {
      int n0 = nt * 16;
      v8f acc = {};
      for (int k0 = 0; k0 < 128; k0 += 32)
        acc = wmma_f16(load_a16(sXt + k0, RH),
                       load_bT16(wih + n0 * RH + k0, RH), acc);
      if (n0 < 256) {
        // r/z gates: accumulate x@Wih and h@Whh into the same tile
        for (int k0 = 0; k0 < 128; k0 += 32)
          acc = wmma_f16(load_a16(sH16 + k0, RH),
                         load_bT16(whh + n0 * RH + k0, RH), acc);
#pragma unroll
        for (int i = 0; i < 8; ++i)
          sRZ[(mh + i) * 256 + n0 + n] = acc[i] + bih[n0 + n] + bhh[n0 + n];
      } else {
#pragma unroll
        for (int i = 0; i < 8; ++i)
          sIN[(mh + i) * RH + (n0 - 256) + n] = acc[i] + bih[n0 + n];
        v8f acc2 = {};
        for (int k0 = 0; k0 < 128; k0 += 32)
          acc2 = wmma_f16(load_a16(sH16 + k0, RH),
                          load_bT16(whh + n0 * RH + k0, RH), acc2);
#pragma unroll
        for (int i = 0; i < 8; ++i)
          sHN[(mh + i) * RH + (n0 - 256) + n] = acc2[i] + bhh[n0 + n];
      }
    }
    __syncthreads();

    for (int i = tid; i < NAg * RH; i += 256) {
      int m = i >> 7, k = i & 127;
      float r  = 1.f / (1.f + expf(-sRZ[m * 256 + k]));
      float z  = 1.f / (1.f + expf(-sRZ[m * 256 + 128 + k]));
      float nn = tanhf(sIN[i] + r * sHN[i]);
      float hv = (1.f - z) * nn + z * sH[i];
      sH[i] = hv; sH16[i] = (F16)hv;
      size_t o = (size_t)e * NAg * RH + i;
      hs_out[o] = hv; hsh[o] = (F16)hv;
    }
    __syncthreads();
  }
}

// -------- K3/K5: 16-entity attention; mode 0 = gat, mode 1 = sp1 (+ELU + fused fc3 -> q) ------
__global__ void __launch_bounds__(128) k_attn16(
    const F16* __restrict__ hsh, const unsigned char* __restrict__ em,
    const unsigned char* __restrict__ adjb,
    const F16* __restrict__ win, const F16* __restrict__ wout, const float* __restrict__ bout,
    int mode, F16* __restrict__ outg,
    const F16* __restrict__ wfc3, const float* __restrict__ bfc3, float* __restrict__ qout)
{
  __shared__ alignas(16) char smem[49152];
  F16*   sH   = (F16*)smem;               // 16x128
  F16*   sQ   = (F16*)(smem + 4096);
  F16*   sK   = (F16*)(smem + 8192);
  F16*   sV   = (F16*)(smem + 12288);
  float* sLf  = (float*)(smem + 16384);   // 4 x 16x16
  F16*   sLh  = (F16*)(smem + 20480);     // 4 x 16x32 (K zero-padded to 32)
  F16*   sVp  = (F16*)(smem + 24576);     // 4 x 32x32 (rows 16..31 zero)
  float* sO   = (float*)(smem + 32768);   // 16x128
  F16*   sO16 = (F16*)(smem + 40960);
  F16*   sM16 = (F16*)(smem + 45056);     // attn-block output tile (both modes)
  const int tid = threadIdx.x, lane = tid & 31, wv = tid >> 5;
  const int e = blockIdx.x;
  const int n = lane & 15, mh = (lane & 16) ? 8 : 0;
  const unsigned char* emr = em + (size_t)e * NE;

  {
    const char* g = (const char*)(hsh + (size_t)e * NAg * RH);
    for (int c = tid; c < 256; c += 128)
      g2l_b128(g + c * 16, (char*)sH + c * 16);
    async_wait0();
  }
  __syncthreads();

  for (int t = wv; t < 24; t += 4) {
    int n0g = t * 16; F16* dst; int dcol;
    if (t < 8)       { dst = sQ; dcol = n0g; }
    else if (t < 16) { dst = sK; dcol = n0g - 128; }
    else             { dst = sV; dcol = n0g - 256; }
    v8f acc = {};
    for (int k0 = 0; k0 < 128; k0 += 32)
      acc = wmma_f16(load_a16(sH + k0, RH),
                     load_bT16(win + n0g * RH + k0, RH), acc);
#pragma unroll
    for (int i = 0; i < 8; ++i) dst[(mh + i) * RH + dcol + n] = (F16)acc[i];
  }
  __syncthreads();

  // logits (16x16 per head)
  {
    int h = wv;
    v8f acc = {};
    acc = wmma_f16(load_a16(sQ + h * 32, RH), load_bT16(sK + h * 32, RH), acc);
    const unsigned char* adjr = mode ? (adjb + (size_t)e * NAg * NAg) : (const unsigned char*)0;
#pragma unroll
    for (int i = 0; i < 8; ++i) {
      int m = mh + i, col = n;
      float v = acc[i] * 0.17677669529663689f;
      bool blk = mode ? (adjr[m * NAg + col] != 0) : (emr[m] || emr[col]);
      if (blk) v = -3.0e38f;
      sLf[h * 256 + m * 16 + col] = v;
    }
  }
  // padded V per head: rows >= 16 are zero (K padded 16 -> 32)
  for (int i = tid; i < 4096; i += 128) {
    int h = i >> 10, r = i & 1023, kk = r >> 5, nn = r & 31;
    sVp[i] = (kk < 16) ? sV[kk * RH + h * 32 + nn] : (F16)0.0f;
  }
  __syncthreads();
  if (lane < 16) {
    float* row  = sLf + wv * 256 + lane * 16;
    F16*   wrow = sLh + wv * 512 + lane * 32;
    float mx = -3.0e38f;
    for (int j = 0; j < 16; ++j) mx = fmaxf(mx, row[j]);
    if (mx > -1.0e38f) {
      float s = 0.f;
      for (int j = 0; j < 16; ++j) s += expf(row[j] - mx);
      float inv = 1.f / s;
      for (int j = 0; j < 16; ++j) wrow[j] = (F16)(expf(row[j] - mx) * inv);
    } else {
      for (int j = 0; j < 16; ++j) wrow[j] = (F16)0.f;
    }
    for (int j = 16; j < 32; ++j) wrow[j] = (F16)0.f;
  }
  __syncthreads();
  {
    int h = wv;
    for (int nt = 0; nt < 2; ++nt) {
      int n0 = nt * 16;
      v8f acc = {};
      acc = wmma_f16(load_a16(sLh + h * 512, 32), load_b16(sVp + h * 1024 + n0, 32), acc);
#pragma unroll
      for (int i = 0; i < 8; ++i) sO[(mh + i) * RH + h * 32 + n0 + n] = acc[i];
    }
  }
  __syncthreads();
  for (int i = tid; i < NAg * RH; i += 128) sO16[i] = (F16)sO[i];
  __syncthreads();

  for (int nt = wv; nt < 8; nt += 4) {
    int n0 = nt * 16;
    v8f acc = {};
    for (int k0 = 0; k0 < 128; k0 += 32)
      acc = wmma_f16(load_a16(sO16 + k0, RH),
                     load_bT16(wout + n0 * RH + k0, RH), acc);
#pragma unroll
    for (int i = 0; i < 8; ++i) {
      int m = mh + i;
      float v = acc[i] + bout[n0 + n];
      if (emr[m]) v = 0.f;
      if (mode) v = (v > 0.f) ? v : (expf(v) - 1.f);       // ELU (sp1 path)
      sM16[m * RH + n0 + n] = (F16)v;
    }
  }
  __syncthreads();
  if (mode) {
    if (wv == 0) {  // fc3 on [hs, msg] (K=256), single N tile of 16
      v8f acc = {};
      for (int k0 = 0; k0 < 256; k0 += 32) {
        v16h a = (k0 < 128) ? load_a16(sH + k0, RH) : load_a16(sM16 + (k0 - 128), RH);
        acc = wmma_f16(a, load_bT16(wfc3 + k0, 256), acc);
      }
#pragma unroll
      for (int i = 0; i < 8; ++i) {
        int m = mh + i;
        float v = acc[i] + bfc3[n];
        if (emr[m]) v = 0.f;
        qout[((size_t)e * NAg + m) * NACT + n] = v;
      }
    }
  } else {
    // drain gat_state tile to global, coalesced/async
    char* g = (char*)(outg + (size_t)e * NAg * RH);
    const char* l = (const char*)sM16;
    for (int c = tid; c < 256; c += 128)
      l2g_b128(g + c * 16, l + c * 16);
  }
}

// ---------------- K4: pairwise gumbel scheduler MLP -> adjacency block mask -------------------
__global__ void __launch_bounds__(128) k4_sched(
    const F16* __restrict__ gath, const unsigned char* __restrict__ em,
    const F16* __restrict__ w1, const float* __restrict__ b1,
    const F16* __restrict__ w2, const float* __restrict__ b2,
    const F16* __restrict__ w3, const float* __restrict__ b3,
    unsigned char* __restrict__ adjb)
{
  __shared__ alignas(16) char smem[55296];
  F16*   g16 = (F16*)smem;              // 16x128
  F16*   sA  = (F16*)(smem + 4096);     // 4 waves x 16x256
  F16*   sH1 = (F16*)(smem + 36864);    // 4 x 16x64
  F16*   sH2 = (F16*)(smem + 45056);    // 4 x 16x64
  float* sP  = (float*)(smem + 53248);  // 4 x 16x2
  const int tid = threadIdx.x, lane = tid & 31, wv = tid >> 5;
  const int e = blockIdx.x;
  const int n = lane & 15, mh = (lane & 16) ? 8 : 0;
  const unsigned char* emr = em + (size_t)e * NE;

  {
    const char* g = (const char*)(gath + (size_t)e * NAg * RH);
    for (int c = tid; c < 256; c += 128)
      g2l_b128(g + c * 16, (char*)g16 + c * 16);
    async_wait0();
  }
  __syncthreads();

  F16*   sAw  = sA  + wv * 4096;
  F16*   sH1w = sH1 + wv * 1024;
  F16*   sH2w = sH2 + wv * 1024;
  float* sPw  = sP  + wv * 32;

  for (int it = 0; it < 4; ++it) {
    int mt = wv + it * 4;  // tile mt covers pairs (i=mt, j=0..15)
    for (int idx = lane; idx < 4096; idx += 32) {
      int r = idx >> 8, k = idx & 255;
      sAw[idx] = (k < 128) ? g16[mt * RH + k] : g16[r * RH + (k - 128)];
    }
    __syncthreads();
    for (int nt = 0; nt < 4; ++nt) {   // h1 = relu(pair @ W1^T + b1), N=64, K=256
      int n0 = nt * 16;
      v8f acc = {};
      for (int k0 = 0; k0 < 256; k0 += 32)
        acc = wmma_f16(load_a16(sAw + k0, 256), load_bT16(w1 + n0 * 256 + k0, 256), acc);
#pragma unroll
      for (int i = 0; i < 8; ++i)
        sH1w[(mh + i) * 64 + n0 + n] = (F16)fmaxf(acc[i] + b1[n0 + n], 0.f);
    }
    __syncthreads();
    for (int nt = 0; nt < 4; ++nt) {   // h2 = relu(h1 @ W2^T + b2), N=64, K=64
      int n0 = nt * 16;
      v8f acc = {};
      for (int k0 = 0; k0 < 64; k0 += 32)
        acc = wmma_f16(load_a16(sH1w + k0, 64), load_bT16(w2 + n0 * 64 + k0, 64), acc);
#pragma unroll
      for (int i = 0; i < 8; ++i)
        sH2w[(mh + i) * 64 + n0 + n] = (F16)fmaxf(acc[i] + b2[n0 + n], 0.f);
    }
    __syncthreads();
    {  // 2-class logits + gumbel perturbation (lane = (row, class))
      int r = lane & 15, cls = lane >> 4;
      float d = b3[cls];
      for (int k = 0; k < 64; ++k)
        d += (float)sH2w[r * 64 + k] * (float)w3[cls * 64 + k];
      unsigned long long key =
          ((unsigned long long)((e * NAg + mt) * NAg + r) << 1) | (unsigned)cls;
      float u = u01hash(key * 0x9E3779B97F4A7C15ULL + 0x0123456789ABCDEFULL);
      sPw[r * 2 + cls] = d - logf(-logf(u));
    }
    __syncthreads();
    if (lane < 16) {
      int r = lane;
      bool fom   = (emr[mt] != 0) || (emr[r] != 0);
      bool hard1 = sPw[r * 2 + 1] > sPw[r * 2 + 0];
      // adj = fom ? 1 : 1-hard1 ; sp1 pre_mask blocks where adj > 0.5
      adjb[(size_t)e * NAg * NAg + mt * NAg + r] = (fom || !hard1) ? 1 : 0;
    }
    __syncthreads();
  }
}

extern "C" void kernel_launch(void* const* d_in, const int* in_sizes, int n_in,
                              void* d_out, int out_size, void* d_ws, size_t ws_size,
                              hipStream_t stream)
{
  (void)in_sizes; (void)n_in; (void)out_size; (void)ws_size;
  const float* ents = (const float*)d_in[0];
  const float* h0   = (const float*)d_in[1];
  const unsigned char* om = (const unsigned char*)d_in[2];
  const unsigned char* em = (const unsigned char*)d_in[3];
  const float* fc1_W = (const float*)d_in[4];
  const float* fc1_b = (const float*)d_in[5];
  const float* awin  = (const float*)d_in[6];
  const float* awout = (const float*)d_in[7];
  const float* about = (const float*)d_in[8];
  const float* fc2_W = (const float*)d_in[9];
  const float* fc2_b = (const float*)d_in[10];
  const float* gwih  = (const float*)d_in[11];
  const float* gwhh  = (const float*)d_in[12];
  const float* gbih  = (const float*)d_in[13];
  const float* gbhh  = (const float*)d_in[14];
  const float* gwin  = (const float*)d_in[15];
  const float* gwout = (const float*)d_in[16];
  const float* gbout = (const float*)d_in[17];
  const float* m1w1  = (const float*)d_in[18];
  const float* m1b1  = (const float*)d_in[19];
  const float* m1w2  = (const float*)d_in[20];
  const float* m1b2  = (const float*)d_in[21];
  const float* m1w3  = (const float*)d_in[22];
  const float* m1b3  = (const float*)d_in[23];
  const float* swin  = (const float*)d_in[24];
  const float* swout = (const float*)d_in[25];
  const float* sbout = (const float*)d_in[26];
  const float* fc3_W = (const float*)d_in[27];
  const float* fc3_b = (const float*)d_in[28];

  char* ws = (char*)d_ws;
  F16* wh_fc1   = (F16*)ws;
  F16* wh_awin  = wh_fc1   + 12288;
  F16* wh_awout = wh_awin  + 49152;
  F16* wh_fc2   = wh_awout + 16384;
  F16* wh_wih   = wh_fc2   + 16384;
  F16* wh_whh   = wh_wih   + 49152;
  F16* wh_gwin  = wh_whh   + 49152;
  F16* wh_gwout = wh_gwin  + 49152;
  F16* wh_swin  = wh_gwout + 16384;
  F16* wh_swout = wh_swin  + 49152;
  F16* wh_m1    = wh_swout + 16384;
  F16* wh_m2    = wh_m1    + 16384;
  F16* wh_m3    = wh_m2    + 4096;
  F16* wh_fc3   = wh_m3    + 128;

  F16* x3h  = (F16*)(ws + (1 << 20));
  F16* hsh  = x3h + (size_t)BSTS * NAg * RH;
  F16* gath = hsh + (size_t)BSTS * NAg * RH;
  unsigned char* adjb = (unsigned char*)(gath + (size_t)BSTS * NAg * RH);

  float* qout   = (float*)d_out;                          // (bs,ts,NA,NACT)
  float* hs_out = qout + (size_t)BSTS * NAg * NACT;       // (bs*ts,NA,RH)

  auto cv = [&](const float* s, F16* d, int ne) {
    cvt_f32_to_f16<<<(ne + 255) / 256, 256, 0, stream>>>(s, d, ne);
  };
  cv(fc1_W, wh_fc1, 12288);  cv(awin, wh_awin, 49152);  cv(awout, wh_awout, 16384);
  cv(fc2_W, wh_fc2, 16384);  cv(gwih, wh_wih, 49152);   cv(gwhh, wh_whh, 49152);
  cv(gwin, wh_gwin, 49152);  cv(gwout, wh_gwout, 16384);
  cv(swin, wh_swin, 49152);  cv(swout, wh_swout, 16384);
  cv(m1w1, wh_m1, 16384);    cv(m1w2, wh_m2, 4096);     cv(m1w3, wh_m3, 128);
  cv(fc3_W, wh_fc3, 4096);

  k1_fc1_attn_fc2<<<BSTS, 128, 0, stream>>>(ents, om, em, wh_fc1, fc1_b, wh_awin,
                                            wh_awout, about, wh_fc2, fc2_b, x3h);
  k2_gru<<<BS, 256, 0, stream>>>(h0, x3h, wh_wih, wh_whh, gbih, gbhh, hs_out, hsh);
  k_attn16<<<BSTS, 128, 0, stream>>>(hsh, em, (const unsigned char*)0,
                                     wh_gwin, wh_gwout, gbout, 0, gath,
                                     wh_fc3, fc3_b, qout);
  k4_sched<<<BSTS, 128, 0, stream>>>(gath, em, wh_m1, m1b1, wh_m2, m1b2, wh_m3, m1b3, adjb);
  k_attn16<<<BSTS, 128, 0, stream>>>(hsh, em, adjb, wh_swin, wh_swout, sbout, 1,
                                     (F16*)0, wh_fc3, fc3_b, qout);
}
